// SparseConvNeXtBlock3d_44839458570626
// MI455X (gfx1250) — compile-verified, matrix operand order checked by reading, not run on hardware
//
#include <hip/hip_runtime.h>

typedef __attribute__((ext_vector_type(16))) __bf16 v16bf;
typedef __attribute__((ext_vector_type(8)))  float  v8f;
typedef __attribute__((ext_vector_type(4)))  unsigned int v4u;
typedef __attribute__((ext_vector_type(8)))  int v8i;
typedef __attribute__((ext_vector_type(4)))  int v4i;

#define NTAP 27
#define LNS 68                       // padded LDS row stride (floats)
#define TILE_ELEMS 512               // one 32x16 bf16 B-fragment tile
#define WCONV_TILES (NTAP * 8)       // 27 taps * (2 kchunks * 4 ntiles)
#define W1_TILES 32                  // 2 kchunks * 16 ntiles
#define W2_TILES 32                  // 8 kchunks * 4 ntiles
#define TOTAL_TILES (WCONV_TILES + W1_TILES + W2_TILES)
#define WFRAG_BYTES ((size_t)TOTAL_TILES * TILE_ELEMS * 2)   // 286,720 B

union BFV  { v16bf v; unsigned short u[16]; unsigned int d[8]; };

__device__ __forceinline__ unsigned short f2bf(float f) {
  unsigned int u = __float_as_uint(f);
  u += 0x7FFFu + ((u >> 16) & 1u);           // round-to-nearest-even
  return (unsigned short)(u >> 16);
}

__device__ __forceinline__ v8f zero8() {
  v8f z = {0.f, 0.f, 0.f, 0.f, 0.f, 0.f, 0.f, 0.f};
  return z;
}

__device__ __forceinline__ v16bf zerobf() {
  BFV r;
  #pragma unroll
  for (int i = 0; i < 8; ++i) r.d[i] = 0;
  return r.v;
}

__device__ __forceinline__ v16bf frag_from(uint4 lo, uint4 hi) {
  BFV r;
  r.d[0] = lo.x; r.d[1] = lo.y; r.d[2] = lo.z; r.d[3] = lo.w;
  r.d[4] = hi.x; r.d[5] = hi.y; r.d[6] = hi.z; r.d[7] = hi.w;
  return r.v;
}

__device__ __forceinline__ v16bf packA(float4 a, float4 b, float4 c, float4 d) {
  BFV r;
  r.u[0] = f2bf(a.x);  r.u[1] = f2bf(a.y);  r.u[2] = f2bf(a.z);  r.u[3] = f2bf(a.w);
  r.u[4] = f2bf(b.x);  r.u[5] = f2bf(b.y);  r.u[6] = f2bf(b.z);  r.u[7] = f2bf(b.w);
  r.u[8] = f2bf(c.x);  r.u[9] = f2bf(c.y);  r.u[10] = f2bf(c.z); r.u[11] = f2bf(c.w);
  r.u[12] = f2bf(d.x); r.u[13] = f2bf(d.y); r.u[14] = f2bf(d.z); r.u[15] = f2bf(d.w);
  return r.v;
}

__device__ __forceinline__ float half_reduce(float x) {
  x += __shfl_xor(x, 1, 32);
  x += __shfl_xor(x, 2, 32);
  x += __shfl_xor(x, 4, 32);
  x += __shfl_xor(x, 8, 32);
  return x;
}

// ---------------------------------------------------------------------------
// TDM: DMA an 8KB weight block (4096 bf16, contiguous) from global to LDS.
// D# built per CDNA5 ISA 8.3/8.4: degenerate 2D tensor 4096x1, data_size=2B.
// This toolchain exposes the 6-arg builtin:
//   (uint32x4 g0, int32x8 g1, int32x4 g2, int32x4 g3, int32x8 pad, i32 cpol)
// ---------------------------------------------------------------------------
#if __has_builtin(__builtin_amdgcn_tensor_load_to_lds)
#define HAVE_TDM 1
__device__ __forceinline__ void tdm_load_8kb(const unsigned short* src,
                                             unsigned lds_off) {
  unsigned long long ga = (unsigned long long)(uintptr_t)src;
  v4u g0;
  g0.x = 1u;                                          // count=1, user descriptor
  g0.y = lds_off;                                     // lds_addr (bytes)
  g0.z = (unsigned)(ga & 0xFFFFFFFFu);                // global_addr[31:0]
  g0.w = (unsigned)((ga >> 32) & 0x01FFFFFFu) | (2u << 30);  // addr[56:32]|type=2
  v8i g1;
  g1[0] = (int)(1u << 16);                 // workgroup_mask=0, data_size=1 (2B)
  g1[1] = (int)(0x1000u << 16);            // tensor_dim0[15:0]=4096 @ bits63:48
  g1[2] = (int)(1u << 16);                 // tensor_dim0 hi=0 | tensor_dim1=1
  g1[3] = (int)(0x1000u << 16);            // tensor_dim1 hi=0 | tile_dim0=4096
  g1[4] = (int)1;                          // tile_dim1=1, tile_dim2=0
  g1[5] = (int)0x1000u;                    // tensor_dim0_stride lo32 = 4096
  g1[6] = (int)(0x1000u << 16);            // d0stride hi=0 | d1stride lo16=4096
  g1[7] = 0;                               // d1stride hi
  v4i z4 = {0, 0, 0, 0};
  v8i z8 = {0, 0, 0, 0, 0, 0, 0, 0};
  __builtin_amdgcn_tensor_load_to_lds(g0, g1, z4, z4, z8, 0);
}
#else
#define HAVE_TDM 0
#endif

// ---------------------------------------------------------------------------
// Pre-kernel 1: swizzle f32 weights into bf16 WMMA B-fragment order.
// Tile = 32x16 (KxN).  lane L: N = L%16, hi = L/16.
// Storage: tile*512 + half16*256 + lane*8 + hsub  (two 16B runs per lane)
//   where K = half16*16 + hi*8 + hsub.
// ---------------------------------------------------------------------------
__global__ void swizzle_weights_kernel(const float* __restrict__ Wc,
                                       const float* __restrict__ W1,
                                       const float* __restrict__ W2,
                                       unsigned short* __restrict__ ws) {
  int gid = blockIdx.x * 256 + threadIdx.x;
  if (gid >= TOTAL_TILES * TILE_ELEMS) return;
  int tile = gid / TILE_ELEMS;
  int e    = gid % TILE_ELEMS;
  int half16 = e >> 8, lane = (e >> 3) & 31, hsub = e & 7;
  int n = lane & 15, hi = lane >> 4;
  int koff = half16 * 16 + hi * 8 + hsub;
  float v;
  if (tile < WCONV_TILES) {
    int k = tile >> 3, rem = tile & 7, kc = rem >> 2, nt = rem & 3;
    v = Wc[k * 4096 + (kc * 32 + koff) * 64 + nt * 16 + n];
  } else if (tile < WCONV_TILES + W1_TILES) {
    int t2 = tile - WCONV_TILES;
    int kc = t2 >> 4, ntile = t2 & 15;
    v = W1[(kc * 32 + koff) * 256 + ntile * 16 + n];
  } else {
    int t3 = tile - WCONV_TILES - W1_TILES;
    int kcg = t3 >> 2, nt = t3 & 3;
    v = W2[(kcg * 32 + koff) * 64 + nt * 16 + n];
  }
  ws[gid] = f2bf(v);
}

// Pre-kernel 2: feats f32 -> bf16 (4 elems / thread)
__global__ void feats_to_bf16_kernel(const float* __restrict__ f,
                                     unsigned short* __restrict__ o,
                                     long long n4) {
  long long i = (long long)blockIdx.x * 256 + threadIdx.x;
  if (i >= n4) return;
  float4 v = ((const float4*)f)[i];
  union { unsigned long long u64; unsigned short u[4]; } r;
  r.u[0] = f2bf(v.x); r.u[1] = f2bf(v.y); r.u[2] = f2bf(v.z); r.u[3] = f2bf(v.w);
  ((unsigned long long*)o)[i] = r.u64;
}

// ---------------------------------------------------------------------------
// neighbor index for tap k of this lane's voxel row
// ---------------------------------------------------------------------------
__device__ __forceinline__ int neighbor_idx(int k, bool vok, int cx, int cy,
                                            int cz, const int* __restrict__ grid,
                                            int R) {
  if (!vok) return -1;
  int dx = k / 9 - 1, dy = (k / 3) % 3 - 1, dz = k % 3 - 1;
  int x = cx + dx, y = cy + dy, z = cz + dz;
  if ((unsigned)x < (unsigned)R && (unsigned)y < (unsigned)R &&
      (unsigned)z < (unsigned)R)
    return grid[(x * R + y) * R + z];
  return -1;
}

__device__ __forceinline__ void load_afrag(int idx,
                                           const float* __restrict__ feats,
                                           const unsigned short* __restrict__ fbf,
                                           int hi, v16bf& a0, v16bf& a1) {
  if (idx >= 0) {
    if (fbf) {
      const unsigned short* f = fbf + (long long)idx * 64;
      const int cA = hi * 8;
      uint4 l0 = *(const uint4*)(f + cA);
      uint4 h0 = *(const uint4*)(f + cA + 16);
      uint4 l1 = *(const uint4*)(f + cA + 32);
      uint4 h1 = *(const uint4*)(f + cA + 48);
      a0 = frag_from(l0, h0);
      a1 = frag_from(l1, h1);
    } else {
      const float* f = feats + (long long)idx * 64;
      const int c0 = hi * 8;
      float4 p0 = *(const float4*)(f + c0);
      float4 p1 = *(const float4*)(f + c0 + 4);
      float4 p2 = *(const float4*)(f + c0 + 16);
      float4 p3 = *(const float4*)(f + c0 + 20);
      float4 q0 = *(const float4*)(f + c0 + 32);
      float4 q1 = *(const float4*)(f + c0 + 36);
      float4 q2 = *(const float4*)(f + c0 + 48);
      float4 q3 = *(const float4*)(f + c0 + 52);
      a0 = packA(p0, p1, p2, p3);
      a1 = packA(q0, q1, q2, q3);
    }
  } else {
    a0 = zerobf();
    a1 = zerobf();
  }
}

// ---------------------------------------------------------------------------
// Main fused kernel.  One wave = 16 voxels x 64 channels; 4 waves / block.
// Conv weights stream into LDS one tap ahead via the Tensor Data Mover.
// ---------------------------------------------------------------------------
__global__ __launch_bounds__(128) void sparse_convnext_kernel(
    const float* __restrict__ feats, const unsigned short* __restrict__ fbf,
    const int* __restrict__ coords, const int* __restrict__ grid,
    const float* __restrict__ bconv, const float* __restrict__ gamma,
    const float* __restrict__ beta, const float* __restrict__ b1,
    const float* __restrict__ b2, const unsigned short* __restrict__ wfrag,
    float* __restrict__ out, int Nvox, int R) {
  __shared__ float lnbuf[4][16 * LNS];
  __shared__ float actbuf[4][16 * LNS];
  __shared__ unsigned short wsmem[2][8 * TILE_ELEMS];   // double-buffered tap wts

  const int lane = threadIdx.x & 31;
  const int w    = threadIdx.x >> 5;
  const int n    = lane & 15;
  const int hi   = lane >> 4;
  const int row  = n;
  const long long vbase = (long long)blockIdx.x * 64 + (long long)w * 16;
  const long long v     = vbase + row;
  const bool vok = (v < Nvox);

  int cx = 0, cy = 0, cz = 0;
  if (vok) {
    cx = coords[v * 3 + 0];
    cy = coords[v * 3 + 1];
    cz = coords[v * 3 + 2];
  }

  // ------- prologue: stage tap-0 weights, load tap-0 A fragments -------
#if HAVE_TDM
  if (w == 0) {
    tdm_load_8kb(wfrag, (unsigned)(uintptr_t)&wsmem[0][0]);
    __builtin_amdgcn_s_wait_tensorcnt(0);
  }
#else
  {
    // cooperative copy fallback: 8KB / 128 threads = 4 x 16B each
    const uint4* src = (const uint4*)wfrag;
    uint4* dst = (uint4*)&wsmem[0][0];
    #pragma unroll
    for (int i = 0; i < 4; ++i)
      dst[threadIdx.x + i * 128] = src[threadIdx.x + i * 128];
  }
#endif
  v16bf a0, a1;
  {
    int idx0 = neighbor_idx(0, vok, cx, cy, cz, grid, R);
    load_afrag(idx0, feats, fbf, hi, a0, a1);
  }
  __syncthreads();

  // ---------------- sparse 3x3x3 conv via WMMA ----------------
  v8f acc[4];
  #pragma unroll
  for (int t = 0; t < 4; ++t) acc[t] = zero8();

  for (int k = 0; k < NTAP; ++k) {
    const int cur = k & 1;
    // 1) start DMA of next tap's weights into the other buffer
#if HAVE_TDM
    if (w == 0 && k + 1 < NTAP)
      tdm_load_8kb(wfrag + (long long)(k + 1) * 8 * TILE_ELEMS,
                   (unsigned)(uintptr_t)&wsmem[cur ^ 1][0]);
#else
    if (k + 1 < NTAP) {
      const uint4* src = (const uint4*)(wfrag + (long long)(k + 1) * 8 * TILE_ELEMS);
      uint4* dst = (uint4*)&wsmem[cur ^ 1][0];
      #pragma unroll
      for (int i = 0; i < 4; ++i)
        dst[threadIdx.x + i * 128] = src[threadIdx.x + i * 128];
    }
#endif
    // 2) preload all 8 B fragments of this tap from LDS
    v16bf b[8];
    {
      const unsigned short* bufk = &wsmem[cur][0];
      #pragma unroll
      for (int j = 0; j < 8; ++j) {
        const unsigned short* tb = bufk + j * TILE_ELEMS;
        uint4 lo = *(const uint4*)(tb + lane * 8);
        uint4 hv = *(const uint4*)(tb + 256 + lane * 8);
        b[j] = frag_from(lo, hv);
      }
    }
    // 3) prefetch next tap's A fragments (global gather hides behind WMMA)
    v16bf na0, na1;
    if (k + 1 < NTAP) {
      int idxn = neighbor_idx(k + 1, vok, cx, cy, cz, grid, R);
      load_afrag(idxn, feats, fbf, hi, na0, na1);
    } else {
      na0 = zerobf();
      na1 = zerobf();
    }
    // 4) 8 back-to-back WMMAs
    #pragma unroll
    for (int t = 0; t < 4; ++t) {
      acc[t] = __builtin_amdgcn_wmma_f32_16x16x32_bf16(
          false, a0, false, b[t], (short)0, acc[t], false, false);
      acc[t] = __builtin_amdgcn_wmma_f32_16x16x32_bf16(
          false, a1, false, b[4 + t], (short)0, acc[t], false, false);
    }
    a0 = na0;
    a1 = na1;
    // 5) make next buffer visible to everyone
#if HAVE_TDM
    if (w == 0) __builtin_amdgcn_s_wait_tensorcnt(0);
#endif
    __syncthreads();
  }

  // conv bias (C/D layout: lane holds channel t*16+n for rows r+8*hi)
  #pragma unroll
  for (int t = 0; t < 4; ++t) {
    float bc = bconv[t * 16 + n];
    #pragma unroll
    for (int r = 0; r < 8; ++r) acc[t][r] += bc;
  }

  // ---------------- LayerNorm over channels (in-register) ----------------
  float mu[8], rs[8];
  #pragma unroll
  for (int r = 0; r < 8; ++r) {
    float s = acc[0][r] + acc[1][r] + acc[2][r] + acc[3][r];
    float q = acc[0][r] * acc[0][r] + acc[1][r] * acc[1][r] +
              acc[2][r] * acc[2][r] + acc[3][r] * acc[3][r];
    s = half_reduce(s);
    q = half_reduce(q);
    float m   = s * (1.0f / 64.0f);
    float var = q * (1.0f / 64.0f) - m * m;
    mu[r] = m;
    rs[r] = rsqrtf(var + 1e-6f);
  }

  float* ln = lnbuf[w];
  #pragma unroll
  for (int t = 0; t < 4; ++t) {
    float g  = gamma[t * 16 + n];
    float bt = beta[t * 16 + n];
    #pragma unroll
    for (int r = 0; r < 8; ++r) {
      float hv = (acc[t][r] - mu[r]) * rs[r] * g + bt;
      ln[(r + 8 * hi) * LNS + t * 16 + n] = hv;
    }
  }
  __syncthreads();

  // ---------------- MLP: 64 -> 256 (SiLU) -> 64 via WMMA ----------------
  const unsigned short* w1f = wfrag + (long long)WCONV_TILES * TILE_ELEMS;
  const unsigned short* w2f = w1f + (long long)W1_TILES * TILE_ELEMS;
  float* act = actbuf[w];

  v8f oacc[4];
  #pragma unroll
  for (int t = 0; t < 4; ++t) oacc[t] = zero8();

  // A fragments of ln are reused by all 4 hidden chunks: load once
  v16bf la[2];
  #pragma unroll
  for (int c = 0; c < 2; ++c) {
    const int c0 = c * 32 + hi * 8;
    const float* lrow = ln + row * LNS;
    float4 p0 = *(const float4*)(lrow + c0);
    float4 p1 = *(const float4*)(lrow + c0 + 4);
    float4 p2 = *(const float4*)(lrow + c0 + 16);
    float4 p3 = *(const float4*)(lrow + c0 + 20);
    la[c] = packA(p0, p1, p2, p3);
  }

  for (int hc = 0; hc < 4; ++hc) {
    v8f hacc[4];
    #pragma unroll
    for (int t = 0; t < 4; ++t) hacc[t] = zero8();

    #pragma unroll
    for (int c = 0; c < 2; ++c) {
      #pragma unroll
      for (int t = 0; t < 4; ++t) {
        int ntile = hc * 4 + t;
        const unsigned short* tb =
            w1f + (long long)(c * 16 + ntile) * TILE_ELEMS;
        uint4 lo = *(const uint4*)(tb + lane * 8);
        uint4 hv = *(const uint4*)(tb + 256 + lane * 8);
        v16bf bfrag = frag_from(lo, hv);
        hacc[t] = __builtin_amdgcn_wmma_f32_16x16x32_bf16(
            false, la[c], false, bfrag, (short)0, hacc[t], false, false);
      }
    }
    __syncthreads();   // actbuf of previous chunk fully consumed

    #pragma unroll
    for (int t = 0; t < 4; ++t) {
      float bb = b1[hc * 64 + t * 16 + n];
      #pragma unroll
      for (int r = 0; r < 8; ++r) {
        float x = hacc[t][r] + bb;
        float s = x / (1.0f + __expf(-x));   // SiLU
        act[(r + 8 * hi) * LNS + t * 16 + n] = s;
      }
    }
    __syncthreads();

    #pragma unroll
    for (int c = 0; c < 2; ++c) {
      const int c0 = c * 32 + hi * 8;
      const float* arow = act + row * LNS;
      float4 p0 = *(const float4*)(arow + c0);
      float4 p1 = *(const float4*)(arow + c0 + 4);
      float4 p2 = *(const float4*)(arow + c0 + 16);
      float4 p3 = *(const float4*)(arow + c0 + 20);
      v16bf a = packA(p0, p1, p2, p3);
      const int kcg = hc * 2 + c;
      #pragma unroll
      for (int t = 0; t < 4; ++t) {
        const unsigned short* tb = w2f + (long long)(kcg * 4 + t) * TILE_ELEMS;
        uint4 lo = *(const uint4*)(tb + lane * 8);
        uint4 hv = *(const uint4*)(tb + 256 + lane * 8);
        v16bf bfrag = frag_from(lo, hv);
        oacc[t] = __builtin_amdgcn_wmma_f32_16x16x32_bf16(
            false, a, false, bfrag, (short)0, oacc[t], false, false);
      }
    }
  }
  __syncthreads();

  // ---------------- stage output (+b2), then coalesced residual+store ----
  #pragma unroll
  for (int t = 0; t < 4; ++t) {
    float bb = b2[t * 16 + n];
    #pragma unroll
    for (int r = 0; r < 8; ++r)
      ln[(r + 8 * hi) * LNS + t * 16 + n] = oacc[t][r] + bb;
  }
  __syncthreads();

  #pragma unroll
  for (int i = 0; i < 8; ++i) {
    int fi = i * 32 + lane;                  // 256 float4 per wave tile
    int r  = fi >> 4;
    int c4 = (fi & 15) * 4;
    long long g = vbase + r;
    if (g < Nvox) {
      float4 s = *(const float4*)(ln + r * LNS + c4);
      float4 f = *(const float4*)(feats + g * 64 + c4);
      float4 o = {s.x + f.x, s.y + f.y, s.z + f.z, s.w + f.w};
      *(float4*)(out + g * 64 + c4) = o;
    }
  }
}

extern "C" void kernel_launch(void* const* d_in, const int* in_sizes, int n_in,
                              void* d_out, int out_size, void* d_ws, size_t ws_size,
                              hipStream_t stream) {
  const float* feats  = (const float*)d_in[0];
  const int*   coords = (const int*)d_in[1];
  const int*   grid   = (const int*)d_in[2];
  const float* Wc     = (const float*)d_in[3];
  const float* bconv  = (const float*)d_in[4];
  const float* gamma  = (const float*)d_in[5];
  const float* beta   = (const float*)d_in[6];
  const float* W1     = (const float*)d_in[7];
  const float* b1     = (const float*)d_in[8];
  const float* W2     = (const float*)d_in[9];
  const float* b2     = (const float*)d_in[10];

  int Nvox = in_sizes[0] / 64;               // feats is [N, 64]
  int R = 1;
  {
    double cr = cbrt((double)in_sizes[2]);   // grid has R^3 elements
    R = (int)(cr + 0.5);
  }

  unsigned short* wfrag = (unsigned short*)d_ws;

  const int swz_total  = TOTAL_TILES * TILE_ELEMS;
  const int swz_blocks = (swz_total + 255) / 256;
  swizzle_weights_kernel<<<swz_blocks, 256, 0, stream>>>(Wc, W1, W2, wfrag);

  // pre-converted bf16 feats if workspace allows (N*64*2 bytes after weights)
  unsigned short* fbf = nullptr;
  size_t fbf_bytes = (size_t)Nvox * 64 * 2;
  if (ws_size >= WFRAG_BYTES + fbf_bytes) {
    fbf = (unsigned short*)((char*)d_ws + WFRAG_BYTES);
    long long n4 = (long long)Nvox * 16;     // float4 count
    int blocks = (int)((n4 + 255) / 256);
    feats_to_bf16_kernel<<<blocks, 256, 0, stream>>>(feats, fbf, n4);
  }

  const int blocks = (Nvox + 63) / 64;
  sparse_convnext_kernel<<<blocks, 128, 0, stream>>>(
      feats, fbf, coords, grid, bconv, gamma, beta, b1, b2, wfrag,
      (float*)d_out, Nvox, R);
}
